// DeepSeekMLA_987842478900
// MI455X (gfx1250) — matrix operand order repaired and checked by reading, not compiled, not run on hardware
//
#include <hip/hip_runtime.h>
#include <hip/hip_bf16.h>
#include <stdint.h>

// ---------------- problem constants ----------------
#define D_MODEL   2048
#define NUM_HEADS 16
#define HEAD_DIM  128
#define D_C       512
#define D_ROT     64
#define BATCH     4
#define SEQ       2048
#define ROWS      (BATCH * SEQ)     // 8192
#define H_DIM     (3 * D_MODEL)     // 6144

typedef __attribute__((ext_vector_type(16))) __bf16 v16bf;
typedef __attribute__((ext_vector_type(8)))  float  v8f;

__device__ __forceinline__ unsigned short f32_to_bf16u(float f) {
  unsigned int u = __float_as_uint(f);
  u += 0x7FFFu + ((u >> 16) & 1u);          // round-to-nearest-even
  return (unsigned short)(u >> 16);
}

// ---------------- Tensor Data Mover (CDNA5 TDM) ----------------
#if defined(__HIP_DEVICE_COMPILE__) && \
    __has_builtin(__builtin_amdgcn_tensor_load_to_lds) && \
    __has_builtin(__builtin_amdgcn_s_wait_tensorcnt)
#define HAVE_TDM 1
#else
#define HAVE_TDM 0
#endif

#if HAVE_TDM
typedef unsigned int u32x4 __attribute__((ext_vector_type(4)));
typedef int          i32x8 __attribute__((ext_vector_type(8)));
typedef int          i32x4 __attribute__((ext_vector_type(4)));

// 2D bf16 tile load: rows x cols tile from global (row stride in elements)
// into LDS at lds_byte_addr, with optional LDS row padding.
// pad_interval_code / pad_amount_code per ISA D# group1 encodings
// (interval = 2^(code+1) DWORDs, amount = code+1 DWORDs); 0xFF = no padding.
__device__ __forceinline__ void tdm_load_2d_bf16(
    unsigned int lds_byte_addr, const void* gsrc,
    unsigned int tile_d0 /*elems per row*/, unsigned int tile_d1 /*rows*/,
    unsigned long long row_stride_elems,
    unsigned int pad_interval_code, unsigned int pad_amount_code) {
  unsigned long long ga = (unsigned long long)(uintptr_t)gsrc;
  u32x4 g0;
  g0[0] = 1u;                                         // count=1 (valid), user mode
  g0[1] = lds_byte_addr;                              // lds_addr
  g0[2] = (unsigned int)(ga & 0xFFFFFFFFu);           // global_addr[31:0]
  g0[3] = (unsigned int)((ga >> 32) & 0x01FFFFFFu)    // global_addr[56:32]
          | (2u << 30);                               // type = 2 ("image")
  i32x8 g1;
  unsigned int w0 = (1u << 16);                       // data_size = 1 -> 2 bytes
  if (pad_interval_code != 0xFFu)
    w0 |= (1u << 20) | (pad_interval_code << 22) | (pad_amount_code << 25);
  g1[0] = (int)w0;
  // bits63:48 = tensor_dim0[15:0] (tensor_dim0 == tile_d0: tile fills tensor row slice)
  g1[1] = (int)((tile_d0 & 0xFFFFu) << 16);
  // bits79:64 = tensor_dim0[31:16]; bits95:80 = tensor_dim1[15:0] (== tile_d1)
  g1[2] = (int)((tile_d0 >> 16) | ((tile_d1 & 0xFFFFu) << 16));
  // bits111:96 = tensor_dim1[31:16]; bits127:112 = tile_dim0
  g1[3] = (int)(((tile_d1 >> 16) & 0xFFFFu) | ((tile_d0 & 0xFFFFu) << 16));
  // bits143:128 = tile_dim1; bits159:144 = tile_dim2 = 0
  g1[4] = (int)(tile_d1 & 0xFFFFu);
  // bits207:160 = tensor_dim0_stride (48b, data_size units)
  g1[5] = (int)(unsigned int)(row_stride_elems & 0xFFFFFFFFull);
  g1[6] = (int)(unsigned int)((row_stride_elems >> 32) & 0xFFFFull); // + dim1_stride[15:0]=0
  g1[7] = 0;
  i32x4 z4 = {0, 0, 0, 0};
  i32x8 z8 = {0, 0, 0, 0, 0, 0, 0, 0};
  // 6-arg form (this toolchain): (g0, g1, g2, g3, g4, cpol); extra groups
  // unused for 2D tensors -> zero-filled.
  __builtin_amdgcn_tensor_load_to_lds(g0, g1, z4, z4, z8, 0);
}
#endif

// ---------------- fp32 -> bf16 conversion ----------------
__global__ void cvt_f32_to_bf16(const float* __restrict__ in,
                                unsigned short* __restrict__ out, int n) {
  int i = blockIdx.x * blockDim.x + threadIdx.x;
  if (i < n) out[i] = f32_to_bf16u(in[i]);
}

// ---------------- rotary on first 64 cols of C (f32, 8192x512) ----------------
__global__ void rotary_kernel(float* __restrict__ C) {
  int idx = blockIdx.x * blockDim.x + threadIdx.x;   // ROWS * 32
  if (idx >= ROWS * 32) return;
  int row = idx >> 5;
  int j   = idx & 31;
  int s   = row & (SEQ - 1);
  float invf = __expf(-(float)j * (9.210340371976184f / 32.0f)); // 10000^(-j/32)
  float ang  = (float)s * invf;
  float c = cosf(ang), sn = sinf(ang);
  float* p = C + (size_t)row * D_C;
  float x1 = p[j], x2 = p[j + 32];
  p[j]      = x1 * c - x2 * sn;
  p[j + 32] = x1 * sn + x2 * c;
}

// ---------------- fragment loaders (ISA 7.12.2 layouts) ----------------
// A (16x32 bf16): lane row = lane&15; K runs {kb..kb+7, kb+16..kb+23}, kb=(lane<16)?0:8
__device__ __forceinline__ v16bf load_frag_a(const unsigned short* base, int ld) {
  int lane = threadIdx.x & 31;
  const unsigned short* p = base + (lane & 15) * ld + ((lane < 16) ? 0 : 8);
  union { v16bf v; uint4 u[2]; } f;
  f.u[0] = *(const uint4*)(p);
  f.u[1] = *(const uint4*)(p + 16);
  return f.v;
}
// B (32x16 bf16) from N-major (transposed) storage: lane col = lane&15,
// K runs {kb..kb+7, kb+16..kb+23}, kb=(lane<16)?0:8
__device__ __forceinline__ v16bf load_frag_b_t(const unsigned short* base, int ld) {
  int lane = threadIdx.x & 31;
  const unsigned short* p = base + (lane & 15) * ld + ((lane < 16) ? 0 : 8);
  union { v16bf v; uint4 u[2]; } f;
  f.u[0] = *(const uint4*)(p);
  f.u[1] = *(const uint4*)(p + 16);
  return f.v;
}

#define WMMA_BF16(A, B, C) \
  __builtin_amdgcn_wmma_f32_16x16x32_bf16(false, (A), false, (B), (short)0, (C), false, false)

// ---------------- tiled WMMA GEMM: D = A(bf16, MxK) * B(bf16, KxN) + bias ----------------
// block = 256 threads (8 wave32), tile 128(M) x 64(N), K step 32.
template <bool OUT_BF16>
__global__ __launch_bounds__(256) void gemm_bf16(
    const unsigned short* __restrict__ A,
    const unsigned short* __restrict__ B,
    const float* __restrict__ bias,
    void* __restrict__ Dout,
    int M, int N, int K) {
  __shared__ __align__(16) unsigned short As[128][40];  // 128 x (32+8)
  __shared__ __align__(16) unsigned short Bs[64][40];   // transposed: [n][k]
  const int m0 = blockIdx.y * 128;
  const int n0 = blockIdx.x * 64;
  const int tid  = threadIdx.x;
  const int lane = tid & 31;
  const int wave = tid >> 5;
  const int wm = (wave & 3) * 32;   // wave M offset within tile
  const int wn = (wave >> 2) * 32;  // wave N offset within tile

  v8f acc[2][2] = {};

  for (int kt = 0; kt < K; kt += 32) {
    __syncthreads();
#if HAVE_TDM
    // A tile 128x32 via Tensor Data Mover (pads 64B rows to 80B in LDS:
    // interval 16 DWORDs -> code 3, pad 4 DWORDs -> code 3)
    if (wave == 0) {
      tdm_load_2d_bf16((unsigned int)(uintptr_t)&As[0][0],
                       A + (size_t)m0 * K + kt, 32u, 128u,
                       (unsigned long long)K, 3u, 3u);
    }
#else
#pragma unroll
    for (int i = 0; i < 2; ++i) {
      int c = tid + 256 * i;
      int row = c >> 2;
      int col = (c & 3) * 8;
      *(uint4*)&As[row][col] = *(const uint4*)(A + (size_t)(m0 + row) * K + kt + col);
    }
#endif
    // B tile 32x64, store transposed [n][k] (TDM cannot transpose)
    {
      int kr = tid >> 3;
      int nc = (tid & 7) * 8;
      uint4 v = *(const uint4*)(B + (size_t)(kt + kr) * N + n0 + nc);
      const unsigned short* s = (const unsigned short*)&v;
#pragma unroll
      for (int j = 0; j < 8; ++j) Bs[nc + j][kr] = s[j];
    }
#if HAVE_TDM
    if (wave == 0) __builtin_amdgcn_s_wait_tensorcnt(0);
#endif
    __syncthreads();

    v16bf a0 = load_frag_a(&As[wm][0], 40);
    v16bf a1 = load_frag_a(&As[wm + 16][0], 40);
    v16bf b0 = load_frag_b_t(&Bs[wn][0], 40);
    v16bf b1 = load_frag_b_t(&Bs[wn + 16][0], 40);
    acc[0][0] = WMMA_BF16(a0, b0, acc[0][0]);
    acc[0][1] = WMMA_BF16(a0, b1, acc[0][1]);
    acc[1][0] = WMMA_BF16(a1, b0, acc[1][0]);
    acc[1][1] = WMMA_BF16(a1, b1, acc[1][1]);
  }

  // epilogue: C layout lane 0-15 -> M=r, lane 16-31 -> M=8+r; N = lane&15
#pragma unroll
  for (int i = 0; i < 2; ++i) {
#pragma unroll
    for (int j = 0; j < 2; ++j) {
      int coln = n0 + wn + j * 16 + (lane & 15);
      float bv = bias[coln];
#pragma unroll
      for (int r = 0; r < 8; ++r) {
        int rowm = m0 + wm + i * 16 + ((lane < 16) ? r : 8 + r);
        float val = acc[i][j][r] + bv;
        if (OUT_BF16)
          ((unsigned short*)Dout)[(size_t)rowm * N + coln] = f32_to_bf16u(val);
        else
          ((float*)Dout)[(size_t)rowm * N + coln] = val;
      }
    }
  }
}

// ---------------- flash attention (per batch, head, 128 q-rows) ----------------
// H layout: [ROWS][6144] bf16, q at col h*128, k at 2048+h*128, v at 4096+h*128.
__global__ __launch_bounds__(256) void attn_kernel(
    const unsigned short* __restrict__ H,
    unsigned short* __restrict__ AO) {
  __shared__ __align__(16) unsigned short Ks[64][136];      // [key][dim]
  __shared__ __align__(16) unsigned short Vts[128][72];     // [dim][key]
  __shared__ __align__(16) unsigned short Ps[8][16][72];    // per-wave P tile

  const int b  = blockIdx.z;
  const int h  = blockIdx.y;
  const int qb = blockIdx.x;
  const int tid  = threadIdx.x;
  const int lane = tid & 31;
  const int wave = tid >> 5;

  const size_t rs = (size_t)H_DIM;  // 6144
  const unsigned short* Hb = H + (size_t)b * SEQ * rs;
  const unsigned short* Qp = Hb + h * HEAD_DIM;
  const unsigned short* Kp = Hb + D_MODEL + h * HEAD_DIM;
  const unsigned short* Vp = Hb + 2 * D_MODEL + h * HEAD_DIM;

  const int q0 = qb * 128 + wave * 16;  // this wave's first q row (s index)
  const int kbA = (lane < 16) ? 0 : 8;

  // Q fragments straight from global (A-layout runs are contiguous in memory)
  v16bf qf[4];
  {
    const unsigned short* qrow = Qp + (size_t)(q0 + (lane & 15)) * rs;
#pragma unroll
    for (int d = 0; d < 4; ++d) {
      union { v16bf v; uint4 u[2]; } f;
      const unsigned short* p = qrow + 32 * d + kbA;
      f.u[0] = *(const uint4*)(p);
      f.u[1] = *(const uint4*)(p + 16);
      qf[d] = f.v;
    }
  }

  v8f o[8] = {};
  float m_i[8], l_i[8];
#pragma unroll
  for (int r = 0; r < 8; ++r) { m_i[r] = -1.0e30f; l_i[r] = 0.0f; }

  const float scale = 0.08838834764831845f;  // 1/sqrt(128)

  for (int k0 = 0; k0 < SEQ; k0 += 64) {
    __syncthreads();
#if HAVE_TDM
    // K tile 64x128 via TDM (pads 256B rows to 272B: interval 64 DW -> code 5,
    // pad 4 DW -> code 3)
    if (wave == 0) {
      tdm_load_2d_bf16((unsigned int)(uintptr_t)&Ks[0][0],
                       Kp + (size_t)k0 * rs, 128u, 64u,
                       (unsigned long long)rs, 5u, 3u);
    }
#else
#pragma unroll
    for (int i = 0; i < 4; ++i) {
      int c = tid + 256 * i;           // 0..1023
      int key = c >> 4;
      int dc  = (c & 15) * 8;
      *(uint4*)&Ks[key][dc] = *(const uint4*)(Kp + (size_t)(k0 + key) * rs + dc);
    }
#endif
    // V tile 64x128 -> LDS transposed [dim][key] for PV (TDM cannot transpose)
#pragma unroll
    for (int i = 0; i < 4; ++i) {
      int c = tid + 256 * i;
      int key = c >> 4;
      int dc  = (c & 15) * 8;
      uint4 v = *(const uint4*)(Vp + (size_t)(k0 + key) * rs + dc);
      const unsigned short* s = (const unsigned short*)&v;
#pragma unroll
      for (int j = 0; j < 8; ++j) Vts[dc + j][key] = s[j];
    }
#if HAVE_TDM
    if (wave == 0) __builtin_amdgcn_s_wait_tensorcnt(0);
#endif
    __syncthreads();

    // S = Q (16x128) * K^T (128x64): 4 key-groups x 4 dim-steps
    v8f sc[4] = {};
#pragma unroll
    for (int g = 0; g < 4; ++g) {
#pragma unroll
      for (int d = 0; d < 4; ++d) {
        v16bf bk = load_frag_b_t(&Ks[g * 16][32 * d], 136);
        sc[g] = WMMA_BF16(qf[d], bk, sc[g]);
      }
    }

    // online softmax (row stats live per-lane: r -> row (lane<16 ? r : 8+r))
    float mb[8];
#pragma unroll
    for (int r = 0; r < 8; ++r) {
#pragma unroll
      for (int g = 0; g < 4; ++g) sc[g][r] *= scale;
      mb[r] = fmaxf(fmaxf(sc[0][r], sc[1][r]), fmaxf(sc[2][r], sc[3][r]));
#pragma unroll
      for (int off = 1; off < 16; off <<= 1)
        mb[r] = fmaxf(mb[r], __shfl_xor(mb[r], off, 32));
      float mnew = fmaxf(m_i[r], mb[r]);
      float fac  = __expf(m_i[r] - mnew);
      m_i[r] = mnew;
      float s = 0.0f;
#pragma unroll
      for (int g = 0; g < 4; ++g) {
        float p = __expf(sc[g][r] - mnew);
        sc[g][r] = p;
        s += p;
      }
#pragma unroll
      for (int off = 1; off < 16; off <<= 1) s += __shfl_xor(s, off, 32);
      l_i[r] = l_i[r] * fac + s;
#pragma unroll
      for (int f = 0; f < 8; ++f) o[f][r] *= fac;
    }

    // P (C-layout f32) -> per-wave LDS (bf16) -> A-layout fragments
    {
      int rowh = (lane < 16) ? 0 : 8;
      int nl = lane & 15;
#pragma unroll
      for (int g = 0; g < 4; ++g)
#pragma unroll
        for (int r = 0; r < 8; ++r)
          Ps[wave][rowh + r][g * 16 + nl] = f32_to_bf16u(sc[g][r]);
    }
    asm volatile("s_wait_dscnt 0x0" ::: "memory");  // same-wave LDS RAW

    // O += P (16x64) * V (64x128): 2 key-steps x 8 dim-tiles
#pragma unroll
    for (int ks = 0; ks < 2; ++ks) {
      v16bf pa = load_frag_a(&Ps[wave][0][ks * 32], 72);
#pragma unroll
      for (int f = 0; f < 8; ++f) {
        v16bf vb = load_frag_b_t(&Vts[f * 16][ks * 32], 72);
        o[f] = WMMA_BF16(pa, vb, o[f]);
      }
    }
  }

  // normalize and store (bf16) to attn-out [ROWS][2048]
#pragma unroll
  for (int r = 0; r < 8; ++r) {
    float inv = 1.0f / l_i[r];
    int srow = q0 + ((lane < 16) ? r : 8 + r);
    size_t base = ((size_t)b * SEQ + srow) * (size_t)D_MODEL + h * HEAD_DIM + (lane & 15);
#pragma unroll
    for (int f = 0; f < 8; ++f)
      AO[base + f * 16] = f32_to_bf16u(o[f][r] * inv);
  }
}

// ---------------- launch ----------------
extern "C" void kernel_launch(void* const* d_in, const int* in_sizes, int n_in,
                              void* d_out, int out_size, void* d_ws, size_t ws_size,
                              hipStream_t stream) {
  const float* x   = (const float*)d_in[0];
  const float* W_c = (const float*)d_in[1];
  const float* b_c = (const float*)d_in[2];
  const float* W_d = (const float*)d_in[3];
  const float* b_d = (const float*)d_in[4];
  const float* W_o = (const float*)d_in[5];
  const float* b_o = (const float*)d_in[6];
  float* out = (float*)d_out;

  char* ws = (char*)d_ws;
  unsigned short* x_bf  = (unsigned short*)(ws + 0);          // 8192*2048 bf16
  unsigned short* Wc_bf = (unsigned short*)(ws + 33554432);   // 2048*512
  unsigned short* Wd_bf = (unsigned short*)(ws + 35651584);   // 512*6144
  unsigned short* Wo_bf = (unsigned short*)(ws + 41943040);   // 2048*2048
  float*          C_f   = (float*)         (ws + 50331648);   // 8192*512 f32
  unsigned short* C_bf  = (unsigned short*)(ws + 67108864);   // 8192*512
  unsigned short* H_bf  = (unsigned short*)(ws + 75497472);   // 8192*6144
  unsigned short* AO_bf = (unsigned short*)(ws + 176160768);  // 8192*2048
  // total workspace: 200 MiB

  {
    int n;
    n = ROWS * D_MODEL;
    cvt_f32_to_bf16<<<(n + 255) / 256, 256, 0, stream>>>(x, x_bf, n);
    n = D_MODEL * D_C;
    cvt_f32_to_bf16<<<(n + 255) / 256, 256, 0, stream>>>(W_c, Wc_bf, n);
    n = D_C * H_DIM;
    cvt_f32_to_bf16<<<(n + 255) / 256, 256, 0, stream>>>(W_d, Wd_bf, n);
    n = D_MODEL * D_MODEL;
    cvt_f32_to_bf16<<<(n + 255) / 256, 256, 0, stream>>>(W_o, Wo_bf, n);
  }

  // C = x * W_c + b_c  (f32 out for rotary precision)
  gemm_bf16<false><<<dim3(D_C / 64, ROWS / 128), 256, 0, stream>>>(
      x_bf, Wc_bf, b_c, C_f, ROWS, D_C, D_MODEL);

  rotary_kernel<<<(ROWS * 32 + 255) / 256, 256, 0, stream>>>(C_f);

  {
    int n = ROWS * D_C;
    cvt_f32_to_bf16<<<(n + 255) / 256, 256, 0, stream>>>(C_f, C_bf, n);
  }

  // H = C * W_d + b_d  (bf16 out: packed q,k,v)
  gemm_bf16<true><<<dim3(H_DIM / 64, ROWS / 128), 256, 0, stream>>>(
      C_bf, Wd_bf, b_d, H_bf, ROWS, H_DIM, D_C);

  // attention -> AO (bf16)
  attn_kernel<<<dim3(SEQ / 128, NUM_HEADS, BATCH), 256, 0, stream>>>(H_bf, AO_bf);

  // out = AO * W_o + b_o  (f32)
  gemm_bf16<false><<<dim3(D_MODEL / 64, ROWS / 128), 256, 0, stream>>>(
      AO_bf, Wo_bf, b_o, out, ROWS, D_MODEL, D_MODEL);
}
